// MultiHeadAttentionNlock_25804163514831
// MI455X (gfx1250) — matrix operand order repaired and checked by reading, not compiled, not run on hardware
//
#include <hip/hip_runtime.h>
#include <hip/hip_bf16.h>

// ---------------------------------------------------------------------------
// MHA forward for MI455X (gfx1250, wave32, WMMA bf16 path).
//   B=2, S=2048, D_MODEL=512, H=8, D_K=64; fp32 I/O, bf16 tensor-core math
//   with fp32 accumulation + fp32 flash softmax.
// Pipeline:
//   0) cvt_kernel x7  : one-pass fp32 -> bf16 for q,k,v and 4 weights, so the
//                       GEMM hot loops are pure b128-load + v_wmma.
//   1) proj_kernel x3 : P = X @ W^T + b (bf16 in, bf16 out), head-split;
//                       Q is pre-scaled by 1/sqrt(dk)=0.125 (exact, pow2);
//                       V stored transposed [B,H,DK,S] for contiguous PV frags.
//   2) attn_kernel    : flash attention, 16 query rows per wave, key blocks
//                       of 32, online softmax, ctx -> bf16 [B,S,D].
//   3) oproj_kernel   : out = ctx @ w_o^T + b_o -> fp32.
// Workspace: 12MB (act bf16) + 2MB (w bf16) + 16MB (Qp,Kp,Vt,Ctx) = 30 MB.
// ---------------------------------------------------------------------------

#define DM    512
#define HEADS 8
#define DK    64
#define BATCH 2
#define SEQ   2048
#define MTOT  (BATCH * SEQ) // 4096

typedef __attribute__((ext_vector_type(16))) __bf16 v16bf;
typedef __attribute__((ext_vector_type(8)))  __bf16 v8bf;
typedef __attribute__((ext_vector_type(8)))  float  v8f;
typedef unsigned short u16;

__device__ __forceinline__ u16 f32_bf16(float f) {
  return __builtin_bit_cast(u16, (__bf16)f);   // hardware v_cvt path
}

// D = A*B + C, A:16x32 bf16, B:32x16 bf16, C/D:16x16 f32
__device__ __forceinline__ v8f wmma_bf16(v16bf a, v16bf b, v8f c) {
  return __builtin_amdgcn_wmma_f32_16x16x32_bf16(false, a, false, b,
                                                 (short)0, c, false, false);
}

// --- fragment loaders (row-major bf16 source, ld in elements) ---------------
// A 16x32: lane<16 -> row, K {0..7,16..23}; lane>=16 -> K {8..15,24..31}
__device__ __forceinline__ v16bf fragA_bf(const u16* base, int ld, int m0, int k0) {
  int lane = threadIdx.x & 31;
  const u16* p = base + (long)(m0 + (lane & 15)) * ld + k0 + ((lane >> 4) << 3);
  v8bf lo = *(const v8bf*)(p);
  v8bf hi = *(const v8bf*)(p + 16);
  return __builtin_shufflevector(lo, hi, 0,1,2,3,4,5,6,7,8,9,10,11,12,13,14,15);
}
// B 32x16 (B[k][n] = Src[n][k], Src row-major along k):
// lane = column n, K {0..7,16..23} (lane<16) / {8..15,24..31} (lane>=16)
__device__ __forceinline__ v16bf fragB_bf(const u16* base, int ld, int n0, int k0) {
  int lane = threadIdx.x & 31;
  const u16* p = base + (long)(n0 + (lane & 15)) * ld + k0 + ((lane >> 4) << 3);
  v8bf lo = *(const v8bf*)(p);
  v8bf hi = *(const v8bf*)(p + 16);
  return __builtin_shufflevector(lo, hi, 0,1,2,3,4,5,6,7,8,9,10,11,12,13,14,15);
}

// ---------------------------------------------------------------------------
// Streaming fp32 -> bf16 conversion, 8 elements per thread (32B in, 16B out).
// ---------------------------------------------------------------------------
__global__ void __launch_bounds__(256)
cvt_kernel(const float* __restrict__ src, u16* __restrict__ dst, int n8) {
  int i = blockIdx.x * 256 + threadIdx.x;
  if (i < n8) {
    v8f f = ((const v8f*)src)[i];
    v8bf h;
#pragma unroll
    for (int j = 0; j < 8; ++j) h[j] = (__bf16)f[j];
    ((v8bf*)dst)[i] = h;
  }
}

// ---------------------------------------------------------------------------
// Projection: out = (X @ W^T + bias) * scale, bf16 in/out.
//   vmode 0: out[((b*H+h)*S + s)*DK + d]   (head-split, for Q and K)
//   vmode 1: out[((b*H+h)*DK + d)*S + s]   (head-split transposed, for V)
// Block = 256 thr (8 waves, 4Mx2N), WG tile 128x128; wave tile 32x64
// (2 A-frags + 4 B-frags -> 8 WMMAs per k-step).
// ---------------------------------------------------------------------------
__global__ void __launch_bounds__(256)
proj_kernel(const u16* __restrict__ X, const u16* __restrict__ W,
            const float* __restrict__ bias, u16* __restrict__ out,
            int vmode, float scale) {
  const int wave = threadIdx.x >> 5;
  const int lane = threadIdx.x & 31;
  const int m0 = blockIdx.x * 128 + (wave & 3) * 32;
  const int n0 = blockIdx.y * 128 + (wave >> 2) * 64;
  v8f acc[2][4] = {};
  for (int k0 = 0; k0 < DM; k0 += 32) {
    v16bf a0 = fragA_bf(X, DM, m0, k0);
    v16bf a1 = fragA_bf(X, DM, m0 + 16, k0);
#pragma unroll
    for (int nt = 0; nt < 4; ++nt) {
      v16bf b = fragB_bf(W, DM, n0 + nt * 16, k0);
      acc[0][nt] = wmma_bf16(a0, b, acc[0][nt]);
      acc[1][nt] = wmma_bf16(a1, b, acc[1][nt]);
    }
  }
  const int half = lane >> 4, nl = lane & 15;
#pragma unroll
  for (int mt = 0; mt < 2; ++mt)
#pragma unroll
    for (int nt = 0; nt < 4; ++nt) {
      int ncol = n0 + nt * 16 + nl;
      float bv = bias[ncol];
      int h = ncol >> 6, d = ncol & 63;
#pragma unroll
      for (int i = 0; i < 8; ++i) {
        int m = m0 + mt * 16 + i + half * 8;     // C layout: row = i + 8*half
        int b = m >> 11, s = m & (SEQ - 1);
        float v = (acc[mt][nt][i] + bv) * scale;
        long off = (vmode == 0)
                     ? ((long)((b * HEADS + h) * SEQ + s)) * DK + d
                     : ((long)((b * HEADS + h) * DK + d)) * SEQ + s;
        out[off] = f32_bf16(v);
      }
    }
}

// ---------------------------------------------------------------------------
// Flash attention: each wave owns one 16-row Q tile of one (b,h); streams
// keys in blocks of 32; online softmax in f32; P routed through wave-private
// LDS to convert C-layout -> A-layout for the PV WMMA. Q already pre-scaled.
// ---------------------------------------------------------------------------
__global__ void __launch_bounds__(256)
attn_kernel(const u16* __restrict__ Qp, const u16* __restrict__ Kp,
            const u16* __restrict__ Vt, u16* __restrict__ Ctx) {
  __shared__ u16 ldsP[8][16][32];      // 1 KB per wave
  const int wave = threadIdx.x >> 5;
  const int lane = threadIdx.x & 31;
  const int task = blockIdx.x * 8 + wave;   // 2048 tasks total
  const int bh = task >> 7;                 // (b,h) in 0..15
  const int qt = task & 127;                // q tile within (b,h)
  const int b = bh >> 3, h = bh & 7;
  const u16* Q = Qp + (long)bh * SEQ * DK;
  const u16* K = Kp + (long)bh * SEQ * DK;
  const u16* V = Vt + (long)bh * DK * SEQ;  // [DK][SEQ]
  const int q0 = qt * 16;

  v16bf aq0 = fragA_bf(Q, DK, q0, 0);
  v16bf aq1 = fragA_bf(Q, DK, q0, 32);
  v8f o[4] = {};
  float mrow[8], lrow[8];
#pragma unroll
  for (int i = 0; i < 8; ++i) { mrow[i] = -3.0e38f; lrow[i] = 0.0f; }
  u16* P = &ldsP[wave][0][0];
  const int half = lane >> 4, nl = lane & 15;

  for (int kb = 0; kb < SEQ; kb += 32) {
    v8f s0 = {}, s1 = {};
    s0 = wmma_bf16(aq0, fragB_bf(K, DK, kb, 0), s0);
    s0 = wmma_bf16(aq1, fragB_bf(K, DK, kb, 32), s0);
    s1 = wmma_bf16(aq0, fragB_bf(K, DK, kb + 16, 0), s1);
    s1 = wmma_bf16(aq1, fragB_bf(K, DK, kb + 16, 32), s1);
#pragma unroll
    for (int i = 0; i < 8; ++i) {
      float x0 = s0[i], x1 = s1[i];      // scale already folded into Q
      // row max across the 16 columns (lanes within each 16-lane half)
      float mx = fmaxf(x0, x1);
      mx = fmaxf(mx, __shfl_xor(mx, 1));
      mx = fmaxf(mx, __shfl_xor(mx, 2));
      mx = fmaxf(mx, __shfl_xor(mx, 4));
      mx = fmaxf(mx, __shfl_xor(mx, 8));
      float mnew = fmaxf(mrow[i], mx);
      float corr = __expf(mrow[i] - mnew);
      mrow[i] = mnew;
      float p0 = __expf(x0 - mnew), p1 = __expf(x1 - mnew);
      float ps = p0 + p1;
      ps += __shfl_xor(ps, 1);
      ps += __shfl_xor(ps, 2);
      ps += __shfl_xor(ps, 4);
      ps += __shfl_xor(ps, 8);
      lrow[i] = lrow[i] * corr + ps;
#pragma unroll
      for (int t = 0; t < 4; ++t) o[t][i] *= corr;
      int row = i + half * 8;                 // C layout
      P[row * 32 + nl] = f32_bf16(p0);
      P[row * 32 + nl + 16] = f32_bf16(p1);
    }
    asm volatile("s_wait_dscnt 0" ::: "memory");   // LDS RAW within the wave
    v16bf ap = fragA_bf(P, 32, 0, 0);              // P as A-frag (16x32 keys)
#pragma unroll
    for (int t = 0; t < 4; ++t)
      o[t] = wmma_bf16(ap, fragB_bf(V, SEQ, t * 16, kb), o[t]);
  }
  // normalize and store ctx (merged heads, bf16)
#pragma unroll
  for (int i = 0; i < 8; ++i) {
    float inv = 1.0f / lrow[i];
    int s = q0 + i + half * 8;
#pragma unroll
    for (int t = 0; t < 4; ++t) {
      int d = t * 16 + nl;
      Ctx[((long)(b * SEQ + s)) * DM + h * DK + d] = f32_bf16(o[t][i] * inv);
    }
  }
}

// ---------------------------------------------------------------------------
// Output projection: out = Ctx(bf16) @ w_o^T(bf16) + b_o -> fp32
// Same tiling as proj_kernel.
// ---------------------------------------------------------------------------
__global__ void __launch_bounds__(256)
oproj_kernel(const u16* __restrict__ Xbf, const u16* __restrict__ W,
             const float* __restrict__ bias, float* __restrict__ out) {
  const int wave = threadIdx.x >> 5;
  const int lane = threadIdx.x & 31;
  const int m0 = blockIdx.x * 128 + (wave & 3) * 32;
  const int n0 = blockIdx.y * 128 + (wave >> 2) * 64;
  v8f acc[2][4] = {};
  for (int k0 = 0; k0 < DM; k0 += 32) {
    v16bf a0 = fragA_bf(Xbf, DM, m0, k0);
    v16bf a1 = fragA_bf(Xbf, DM, m0 + 16, k0);
#pragma unroll
    for (int nt = 0; nt < 4; ++nt) {
      v16bf b = fragB_bf(W, DM, n0 + nt * 16, k0);
      acc[0][nt] = wmma_bf16(a0, b, acc[0][nt]);
      acc[1][nt] = wmma_bf16(a1, b, acc[1][nt]);
    }
  }
  const int half = lane >> 4, nl = lane & 15;
#pragma unroll
  for (int mt = 0; mt < 2; ++mt)
#pragma unroll
    for (int nt = 0; nt < 4; ++nt) {
      int ncol = n0 + nt * 16 + nl;
      float bv = bias[ncol];
#pragma unroll
      for (int i = 0; i < 8; ++i) {
        int m = m0 + mt * 16 + i + half * 8;
        out[(long)m * DM + ncol] = acc[mt][nt][i] + bv;
      }
    }
}

// ---------------------------------------------------------------------------
extern "C" void kernel_launch(void* const* d_in, const int* in_sizes, int n_in,
                              void* d_out, int out_size, void* d_ws, size_t ws_size,
                              hipStream_t stream) {
  // setup_inputs dict order: q, v, k, w_q, b_q, w_k, b_k, w_v, b_v, w_o, b_o
  const float* q   = (const float*)d_in[0];
  const float* v   = (const float*)d_in[1];
  const float* k   = (const float*)d_in[2];
  const float* w_q = (const float*)d_in[3];
  const float* b_q = (const float*)d_in[4];
  const float* w_k = (const float*)d_in[5];
  const float* b_k = (const float*)d_in[6];
  const float* w_v = (const float*)d_in[7];
  const float* b_v = (const float*)d_in[8];
  const float* w_o = (const float*)d_in[9];
  const float* b_o = (const float*)d_in[10];
  float* out = (float*)d_out;

  // workspace layout (bf16 = u16)
  const size_t NA = (size_t)MTOT * DM;   // activation elems (2M)
  const size_t NW = (size_t)DM * DM;     // weight elems (256K)
  u16* Xq  = (u16*)d_ws;       // converted inputs
  u16* Xk  = Xq + NA;
  u16* Xv  = Xk + NA;
  u16* Wq  = Xv + NA;          // converted weights
  u16* Wk  = Wq + NW;
  u16* Wv  = Wk + NW;
  u16* Wo  = Wv + NW;
  u16* Qp  = Wo + NW;          // projected tensors
  u16* Kp  = Qp + NA;
  u16* Vt  = Kp + NA;
  u16* Ctx = Vt + NA;

  // 0) fp32 -> bf16 conversions (8 elems/thread)
  const int act8 = (int)(NA / 8), wgt8 = (int)(NW / 8);
  cvt_kernel<<<act8 / 256, 256, 0, stream>>>(q, Xq, act8);
  cvt_kernel<<<act8 / 256, 256, 0, stream>>>(k, Xk, act8);
  cvt_kernel<<<act8 / 256, 256, 0, stream>>>(v, Xv, act8);
  cvt_kernel<<<wgt8 / 256, 256, 0, stream>>>(w_q, Wq, wgt8);
  cvt_kernel<<<wgt8 / 256, 256, 0, stream>>>(w_k, Wk, wgt8);
  cvt_kernel<<<wgt8 / 256, 256, 0, stream>>>(w_v, Wv, wgt8);
  cvt_kernel<<<wgt8 / 256, 256, 0, stream>>>(w_o, Wo, wgt8);

  // 1) projections (Q pre-scaled by 1/sqrt(DK) = 0.125, exact power of two)
  dim3 gemm_grid(MTOT / 128, DM / 128);       // 32 x 4
  proj_kernel<<<gemm_grid, 256, 0, stream>>>(Xq, Wq, b_q, Qp, 0, 0.125f);
  proj_kernel<<<gemm_grid, 256, 0, stream>>>(Xk, Wk, b_k, Kp, 0, 1.0f);
  proj_kernel<<<gemm_grid, 256, 0, stream>>>(Xv, Wv, b_v, Vt, 1, 1.0f);

  // 2) flash attention: 2048 wave tasks (16 bh x 128 q-tiles), 8 waves/block
  attn_kernel<<<256, 256, 0, stream>>>(Qp, Kp, Vt, Ctx);

  // 3) output projection -> fp32
  oproj_kernel<<<gemm_grid, 256, 0, stream>>>(Ctx, Wo, b_o, out);
}
